// CoHHN_28802050687852
// MI455X (gfx1250) — matrix operand order repaired and checked by reading, not compiled, not run on hardware
//
#include <hip/hip_runtime.h>
#include <hip/hip_bf16.h>

// ---------------------------------------------------------------------------
// CoHHN forward, CDNA5 (gfx1250, wave32).
// Dense inter-gate GEMMs ([N,384]x[384,128]) run on v_wmma_f32_16x16x32_bf16
// with B pre-swizzled into fragment order (one b128-pair load per fragment).
// Sparse spmm scatters + intra-gate softmax aggregation are VALU/atomic bound
// and L2-resident (emb table = 25.6MB << 192MB L2); aggregation amortizes
// emb2 reads across 8 rows per block.
// ---------------------------------------------------------------------------

#define N_NODE  50000
#define N_PRICE 100
#define N_CAT   500
#define EMB     128
#define K3      (3 * EMB)          // 384
#define NNZ_ADJ 1000000
#define LAYERS_FIXED 2             // matches reference constant

typedef __attribute__((ext_vector_type(16))) __bf16 v16bf;
typedef __attribute__((ext_vector_type(8)))  float  v8f;

// ----------------------------- utility kernels -----------------------------

__global__ void zero_f32(float* __restrict__ p, long n) {
    long i = (long)blockIdx.x * blockDim.x + threadIdx.x;
    long stride = (long)gridDim.x * blockDim.x;
    for (; i < n; i += stride) p[i] = 0.0f;
}

__global__ void copy_f32(float* __restrict__ dst, const float* __restrict__ src, long n) {
    long i = (long)blockIdx.x * blockDim.x + threadIdx.x;
    long stride = (long)gridDim.x * blockDim.x;
    for (; i < n; i += stride) dst[i] = src[i];
}

// Combined gate weight, written directly in WMMA B-fragment-swizzled order:
//   logical Wc[k][n] = Wa[n][k] (+ W1[n][k-128] | W2[n][k-256] on K blocks)
//   fragment addr: kt=k/32, kk=k%32, nt=n/16, nn=n%16
//                  lane = (kk&16) + nn, elem = kk&15
//   Wcs[(((kt*8+nt)*32 + lane) << 4) + elem]
// Also bc[n] = ba[n]+b1[n]+b2[n].
__global__ void combine_weights(const float* __restrict__ Wa, const float* __restrict__ ba,
                                const float* __restrict__ W1, const float* __restrict__ b1,
                                const float* __restrict__ W2, const float* __restrict__ b2,
                                __bf16* __restrict__ Wcs, float* __restrict__ bc) {
    int k = blockIdx.x;          // 0..383
    int n = threadIdx.x;         // 0..127
    float w = Wa[n * K3 + k];
    if (k >= EMB && k < 2 * EMB) w += W1[n * EMB + (k - EMB)];
    if (k >= 2 * EMB)            w += W2[n * EMB + (k - 2 * EMB)];
    int kt = k >> 5, kk = k & 31;
    int nt = n >> 4, nn = n & 15;
    int lane = (kk & 16) + nn;
    int elem = kk & 15;
    Wcs[(((kt * 8 + nt) * 32 + lane) << 4) + elem] = (__bf16)w;
    if (k == 0) bc[n] = ba[n] + b1[n] + b2[n];
}

// ----------------------------- sparse spmm ---------------------------------
// out[rows[e]][d] += vals[e] * src[cols[e]][d]   (atomic scatter, 4 dims/thread)
__global__ void spmm_atomic(const int* __restrict__ rows, const int* __restrict__ cols,
                            const float* __restrict__ vals, const float* __restrict__ src,
                            float* __restrict__ out, int nnz) {
    long t = (long)blockIdx.x * blockDim.x + threadIdx.x;
    int e = (int)(t >> 5);
    if (e >= nnz) return;
    int d0 = (int)(t & 31) * 4;
    int r = rows[e], c = cols[e];
    float v = vals[e];
    const float* s = src + (long)c * EMB + d0;
    float* o = out + (long)r * EMB + d0;
    atomicAdd(o + 0, v * s[0]);
    atomicAdd(o + 1, v * s[1]);
    atomicAdd(o + 2, v * s[2]);
    atomicAdd(o + 3, v * s[3]);
}

// ----------------------------- intra-gate ----------------------------------

// s[r] = sum_d src[r][d]   (one 128-thread block per row)
__global__ void rowsum128(const float* __restrict__ src, float* __restrict__ s, int R) {
    __shared__ float red[128];
    int r = blockIdx.x;
    if (r >= R) return;
    red[threadIdx.x] = src[(long)r * EMB + threadIdx.x];
    __syncthreads();
    for (int o = 64; o > 0; o >>= 1) {
        if (threadIdx.x < o) red[threadIdx.x] += red[threadIdx.x + o];
        __syncthreads();
    }
    if (threadIdx.x == 0) s[r] = red[0];
}

// single-block max/min of s -> mm[0]=max, mm[1]=min
__global__ void minmax_f32(const float* __restrict__ s, int n, float* __restrict__ mm) {
    __shared__ float smax[256], smin[256];
    float mx = -3.4e38f, mn = 3.4e38f;
    for (int i = threadIdx.x; i < n; i += blockDim.x) {
        float v = s[i];
        mx = fmaxf(mx, v);
        mn = fminf(mn, v);
    }
    smax[threadIdx.x] = mx; smin[threadIdx.x] = mn;
    __syncthreads();
    for (int o = 128; o > 0; o >>= 1) {
        if (threadIdx.x < o) {
            smax[threadIdx.x] = fmaxf(smax[threadIdx.x], smax[threadIdx.x + o]);
            smin[threadIdx.x] = fminf(smin[threadIdx.x], smin[threadIdx.x + o]);
        }
        __syncthreads();
    }
    if (threadIdx.x == 0) { mm[0] = smax[0]; mm[1] = smin[0]; }
}

// Per row r: M = mv>=0 ? mv*max(s) : mv*min(s)
// den[r] = sum_j exp(mv*s[j]-M)*adj[r][j] + 1e-8 * sum_j exp(mv*s[j]-M)
__global__ void intra_denom(const float* __restrict__ adj, const float* __restrict__ s,
                            const float* __restrict__ mat_v, const float* __restrict__ mm,
                            float* __restrict__ den, float* __restrict__ Mrow,
                            int R, int J) {
    __shared__ float rA[256], rB[256];
    int r = blockIdx.x;
    if (r >= R) return;
    float mv = mat_v[r];
    float M = (mv >= 0.0f) ? mv * mm[0] : mv * mm[1];
    float Sall = 0.0f, Sadj = 0.0f;
    const float* arow = adj + (long)r * J;
    for (int j = threadIdx.x; j < J; j += blockDim.x) {
        float e = __expf(mv * s[j] - M);
        Sall += e;
        Sadj += e * arow[j];
    }
    rA[threadIdx.x] = Sall; rB[threadIdx.x] = Sadj;
    __syncthreads();
    for (int o = 128; o > 0; o >>= 1) {
        if (threadIdx.x < o) {
            rA[threadIdx.x] += rA[threadIdx.x + o];
            rB[threadIdx.x] += rB[threadIdx.x + o];
        }
        __syncthreads();
    }
    if (threadIdx.x == 0) { den[r] = rB[0] + 1e-8f * rA[0]; Mrow[r] = M; }
}

// out[r][d] += sum_{j in chunk} w[r][j] * emb2[j][d]
// 8 rows per block so each emb2 element feeds 8 accumulators.
// grid: (ceil(J/JCHUNK), ceil(R/RPB)); block: 128 threads (thread = dim d)
#define JCHUNK 2048
#define RPB 8
__global__ void intra_aggregate(const float* __restrict__ adj, const float* __restrict__ s,
                                const float* __restrict__ mat_v, const float* __restrict__ Mrow,
                                const float* __restrict__ den, const float* __restrict__ emb2,
                                float* __restrict__ out, int R, int J) {
    __shared__ float wbuf[RPB][128];
    int r0 = blockIdx.y * RPB;
    int j0 = blockIdx.x * JCHUNK;
    int jend = min(J, j0 + JCHUNK);
    int d = threadIdx.x;

    float mv[RPB], M[RPB], invd[RPB];
#pragma unroll
    for (int i = 0; i < RPB; ++i) {
        int r = min(r0 + i, R - 1);
        bool valid = (r0 + i) < R;
        mv[i]   = mat_v[r];
        M[i]    = Mrow[r];
        invd[i] = valid ? (1.0f / den[r]) : 0.0f;   // invd=0 zeroes padded rows
    }

    float acc[RPB];
#pragma unroll
    for (int i = 0; i < RPB; ++i) acc[i] = 0.0f;

    for (int jb = j0; jb < jend; jb += 128) {
        int j = jb + threadIdx.x;
        bool jok = j < jend;
        float sj = jok ? s[j] : 0.0f;
#pragma unroll
        for (int i = 0; i < RPB; ++i) {
            int r = min(r0 + i, R - 1);
            float w = 0.0f;
            if (jok) w = __expf(mv[i] * sj - M[i]) * adj[(long)r * J + j] * invd[i];
            wbuf[i][threadIdx.x] = w;
        }
        __syncthreads();
        int cnt = min(128, jend - jb);
        for (int q = 0; q < cnt; ++q) {
            float v = emb2[(long)(jb + q) * EMB + d];
#pragma unroll
            for (int i = 0; i < RPB; ++i)
                acc[i] = fmaf(wbuf[i][q], v, acc[i]);
        }
        __syncthreads();
    }
#pragma unroll
    for (int i = 0; i < RPB; ++i)
        if (r0 + i < R) atomicAdd(&out[(long)(r0 + i) * EMB + d], acc[i]);
}

// ----------------------------- inter-gate (WMMA) ----------------------------
// gate_pre = [e1|e2|e3] (N,384) @ Wc (384,128) + bc ; g = sigmoid(gate_pre)
// out = e1 + g*e2 + (1-g)*e3 (+ extra)
// Block: 256 threads = 8 waves; each block does a 16-row x 128-col slab.
// Each wave owns one 16x16 output tile; K loop = 384/32 = 12 WMMAs.
// B fragments come pre-swizzled: one contiguous v16bf per (kt, wave, lane).
__global__ void inter_gate_wmma(const float* __restrict__ e1, const float* __restrict__ e2,
                                const float* __restrict__ e3, const float* __restrict__ extra,
                                const __bf16* __restrict__ Wcs, const float* __restrict__ bc,
                                float* __restrict__ out, int n_rows) {
    __shared__ __bf16 Xs[16][K3 + 8];     // 16 rows x 384 cols bf16 (row = 784B, 16B aligned)
    int r0 = blockIdx.x * 16;

    // cooperative stage + f32->bf16 convert (rows clamped => EXEC stays full)
    for (int idx = threadIdx.x; idx < 16 * K3; idx += blockDim.x) {
        int row = idx / K3;
        int col = idx - row * K3;
        int rr = min(r0 + row, n_rows - 1);
        float v;
        if (col < EMB)            v = e1[(long)rr * EMB + col];
        else if (col < 2 * EMB)   v = e2[(long)rr * EMB + (col - EMB)];
        else                      v = e3[(long)rr * EMB + (col - 2 * EMB)];
        Xs[row][col] = (__bf16)v;
    }
    __syncthreads();

    int wave = threadIdx.x >> 5;
    int lane = threadIdx.x & 31;
    int ncol0 = wave * 16;

    int m     = lane & 15;
    int koffA = (lane < 16) ? 0 : 16;     // A frag: contiguous k = 16*(lane/16) + elem (matches Wcs)
    int n     = ncol0 + (lane & 15);

    const v16bf* Wv = (const v16bf*)Wcs;  // fragment-major: [(kt*8+wave)*32 + lane]

    v8f acc = {};
#pragma unroll
    for (int kt = 0; kt < K3 / 32; ++kt) {
        int kbase = kt * 32;
        v16bf a;
#pragma unroll
        for (int i = 0; i < 4; ++i) {
            int k = kbase + koffA + 2 * i;
            a[2 * i]     = Xs[m][k];
            a[2 * i + 1] = Xs[m][k + 1];
            int k2 = kbase + 8 + koffA + 2 * i;
            a[8 + 2 * i]     = Xs[m][k2];
            a[8 + 2 * i + 1] = Xs[m][k2 + 1];
        }
        v16bf b = Wv[(kt * 8 + wave) * 32 + lane];
        acc = __builtin_amdgcn_wmma_f32_16x16x32_bf16(
            false, a, false, b, (short)0, acc, false, false);
    }

    // epilogue: C/D layout -> lane (L&15)=col, VGPR i holds row i + 8*(L>>4)
    int mo = (lane >> 4) * 8;
    float bias = bc[n];
#pragma unroll
    for (int i = 0; i < 8; ++i) {
        int row = r0 + mo + i;
        if (row < n_rows) {
            long off = (long)row * EMB + n;
            float g = 1.0f / (1.0f + __expf(-(acc[i] + bias)));
            float v1 = e1[off], v2 = e2[off], v3 = e3[off];
            float o = v1 + g * v2 + (1.0f - g) * v3;
            if (extra) o += extra[off];
            out[off] = o;
        }
    }
}

// ---------------------------------------------------------------------------

extern "C" void kernel_launch(void* const* d_in, const int* in_sizes, int n_in,
                              void* d_out, int out_size, void* d_ws, size_t ws_size,
                              hipStream_t stream) {
    (void)in_sizes; (void)n_in; (void)out_size; (void)ws_size;

    const int*   adj_rows = (const int*)  d_in[0];
    const int*   adj_cols = (const int*)  d_in[1];
    const float* adj_vals = (const float*)d_in[2];
    const int*   vp_rows  = (const int*)  d_in[3];
    const int*   vp_cols  = (const int*)  d_in[4];
    const float* vp_vals  = (const float*)d_in[5];
    const int*   vc_rows  = (const int*)  d_in[6];
    const int*   vc_cols  = (const int*)  d_in[7];
    const float* vc_vals  = (const float*)d_in[8];
    const float* pv_dense = (const float*)d_in[9];
    const float* pc_dense = (const float*)d_in[10];
    const float* cp_dense = (const float*)d_in[11];
    const float* cv_dense = (const float*)d_in[12];
    const float* embedding= (const float*)d_in[13];
    const float* pri_emb  = (const float*)d_in[14];
    const float* cate_emb = (const float*)d_in[15];
    const float* mat_pv   = (const float*)d_in[16];
    const float* mat_pc   = (const float*)d_in[17];
    const float* mat_cp   = (const float*)d_in[18];
    const float* mat_cv   = (const float*)d_in[19];
    const float* Wai = (const float*)d_in[20]; const float* bai = (const float*)d_in[21];
    const float* Wi1 = (const float*)d_in[22]; const float* bi1 = (const float*)d_in[23];
    const float* Wi2 = (const float*)d_in[24]; const float* bi2 = (const float*)d_in[25];
    const float* Wap = (const float*)d_in[26]; const float* bap = (const float*)d_in[27];
    const float* Wp1 = (const float*)d_in[28]; const float* bp1 = (const float*)d_in[29];
    const float* Wp2 = (const float*)d_in[30]; const float* bp2 = (const float*)d_in[31];
    const float* Wac = (const float*)d_in[32]; const float* bac = (const float*)d_in[33];
    const float* Wc1 = (const float*)d_in[34]; const float* bc1 = (const float*)d_in[35];
    const float* Wc2 = (const float*)d_in[36]; const float* bc2 = (const float*)d_in[37];
    // d_in[38] = layers (device scalar) : fixed to 2 per reference constant.

    // -------- workspace carve-up --------
    char* ws = (char*)d_ws;
    size_t off = 0;
    auto carve = [&](size_t bytes) { char* p = ws + off; off += (bytes + 255) & ~(size_t)255; return p; };

    const long NE = (long)N_NODE * EMB;                 // 6,400,000
    float* e2i     = (float*)carve(NE * 4);
    float* e3i     = (float*)carve(NE * 4);
    float* adj_out = (float*)carve(NE * 4);
    float* embA    = (float*)carve(NE * 4);
    float* embB    = (float*)carve(NE * 4);
    float* priA    = (float*)carve((size_t)N_PRICE * EMB * 4);
    float* priB    = (float*)carve((size_t)N_PRICE * EMB * 4);
    float* catA    = (float*)carve((size_t)N_CAT * EMB * 4);
    float* catB    = (float*)carve((size_t)N_CAT * EMB * 4);
    float* pv_out  = (float*)carve((size_t)N_PRICE * EMB * 4);
    float* pc_out  = (float*)carve((size_t)N_PRICE * EMB * 4);
    float* cp_out  = (float*)carve((size_t)N_CAT * EMB * 4);
    float* cv_out  = (float*)carve((size_t)N_CAT * EMB * 4);
    float* s_item  = (float*)carve((size_t)N_NODE * 4);
    float* s_pri   = (float*)carve((size_t)N_PRICE * 4);
    float* s_cat   = (float*)carve((size_t)N_CAT * 4);
    float* mm_item = (float*)carve(8 * 4);
    float* mm_pri  = (float*)carve(8 * 4);
    float* mm_cat  = (float*)carve(8 * 4);
    float* den     = (float*)carve(512 * 4);
    float* Mrow    = (float*)carve(512 * 4);
    __bf16* Wci    = (__bf16*)carve((size_t)K3 * EMB * 2);
    __bf16* Wcp    = (__bf16*)carve((size_t)K3 * EMB * 2);
    __bf16* Wcc    = (__bf16*)carve((size_t)K3 * EMB * 2);
    float* bci     = (float*)carve(EMB * 4);
    float* bcp     = (float*)carve(EMB * 4);
    float* bcc     = (float*)carve(EMB * 4);

    // -------- constant weight folding (weights do not change across layers) --------
    combine_weights<<<K3, EMB, 0, stream>>>(Wai, bai, Wi1, bi1, Wi2, bi2, Wci, bci);
    combine_weights<<<K3, EMB, 0, stream>>>(Wap, bap, Wp1, bp1, Wp2, bp2, Wcp, bcp);
    combine_weights<<<K3, EMB, 0, stream>>>(Wac, bac, Wc1, bc1, Wc2, bc2, Wcc, bcc);

    const float* emb_src = embedding;
    const float* pri_src = pri_emb;
    const float* cat_src = cate_emb;

    for (int layer = 0; layer < LAYERS_FIXED; ++layer) {
        float* emb_dst = (layer == 0) ? embA : embB;
        float* pri_dst = (layer == 0) ? priA : priB;
        float* cat_dst = (layer == 0) ? catA : catB;

        // ---- item branch sparse inputs ----
        zero_f32<<<4096, 256, 0, stream>>>(e2i, NE);
        zero_f32<<<4096, 256, 0, stream>>>(e3i, NE);
        zero_f32<<<4096, 256, 0, stream>>>(adj_out, NE);
        {
            long t_vp = (long)N_NODE * 32;
            spmm_atomic<<<(int)((t_vp + 255) / 256), 256, 0, stream>>>(
                vp_rows, vp_cols, vp_vals, pri_src, e2i, N_NODE);
            spmm_atomic<<<(int)((t_vp + 255) / 256), 256, 0, stream>>>(
                vc_rows, vc_cols, vc_vals, cat_src, e3i, N_NODE);
            long t_adj = (long)NNZ_ADJ * 32;
            spmm_atomic<<<(int)((t_adj + 255) / 256), 256, 0, stream>>>(
                adj_rows, adj_cols, adj_vals, emb_src, adj_out, NNZ_ADJ);
        }

        // ---- row sums + global min/max for softmax shift ----
        rowsum128<<<N_NODE, 128, 0, stream>>>(emb_src, s_item, N_NODE);
        rowsum128<<<N_PRICE, 128, 0, stream>>>(pri_src, s_pri, N_PRICE);
        rowsum128<<<N_CAT, 128, 0, stream>>>(cat_src, s_cat, N_CAT);
        minmax_f32<<<1, 256, 0, stream>>>(s_item, N_NODE, mm_item);
        minmax_f32<<<1, 256, 0, stream>>>(s_pri, N_PRICE, mm_pri);
        minmax_f32<<<1, 256, 0, stream>>>(s_cat, N_CAT, mm_cat);

        // ---- price branch intra-gates ----
        zero_f32<<<64, 256, 0, stream>>>(pv_out, (long)N_PRICE * EMB);
        intra_denom<<<N_PRICE, 256, 0, stream>>>(pv_dense, s_item, mat_pv, mm_item,
                                                 den, Mrow, N_PRICE, N_NODE);
        {
            dim3 g((N_NODE + JCHUNK - 1) / JCHUNK, (N_PRICE + RPB - 1) / RPB);
            intra_aggregate<<<g, 128, 0, stream>>>(pv_dense, s_item, mat_pv, Mrow, den,
                                                   emb_src, pv_out, N_PRICE, N_NODE);
        }
        zero_f32<<<64, 256, 0, stream>>>(pc_out, (long)N_PRICE * EMB);
        intra_denom<<<N_PRICE, 256, 0, stream>>>(pc_dense, s_cat, mat_pc, mm_cat,
                                                 den, Mrow, N_PRICE, N_CAT);
        {
            dim3 g((N_CAT + JCHUNK - 1) / JCHUNK, (N_PRICE + RPB - 1) / RPB);
            intra_aggregate<<<g, 128, 0, stream>>>(pc_dense, s_cat, mat_pc, Mrow, den,
                                                   cat_src, pc_out, N_PRICE, N_CAT);
        }

        // ---- category branch intra-gates ----
        zero_f32<<<256, 256, 0, stream>>>(cp_out, (long)N_CAT * EMB);
        intra_denom<<<N_CAT, 256, 0, stream>>>(cp_dense, s_pri, mat_cp, mm_pri,
                                               den, Mrow, N_CAT, N_PRICE);
        {
            dim3 g((N_PRICE + JCHUNK - 1) / JCHUNK, (N_CAT + RPB - 1) / RPB);
            intra_aggregate<<<g, 128, 0, stream>>>(cp_dense, s_pri, mat_cp, Mrow, den,
                                                   pri_src, cp_out, N_CAT, N_PRICE);
        }
        zero_f32<<<256, 256, 0, stream>>>(cv_out, (long)N_CAT * EMB);
        intra_denom<<<N_CAT, 256, 0, stream>>>(cv_dense, s_item, mat_cv, mm_item,
                                               den, Mrow, N_CAT, N_NODE);
        {
            dim3 g((N_NODE + JCHUNK - 1) / JCHUNK, (N_CAT + RPB - 1) / RPB);
            intra_aggregate<<<g, 128, 0, stream>>>(cv_dense, s_item, mat_cv, Mrow, den,
                                                   emb_src, cv_out, N_CAT, N_NODE);
        }

        // ---- fused inter-gates (WMMA bf16 GEMM + sigmoid combine) ----
        inter_gate_wmma<<<(N_NODE + 15) / 16, 256, 0, stream>>>(
            emb_src, e2i, e3i, adj_out, Wci, bci, emb_dst, N_NODE);
        inter_gate_wmma<<<(N_PRICE + 15) / 16, 256, 0, stream>>>(
            pri_src, pv_out, pc_out, nullptr, Wcp, bcp, pri_dst, N_PRICE);
        inter_gate_wmma<<<(N_CAT + 15) / 16, 256, 0, stream>>>(
            cat_src, cp_out, cv_out, nullptr, Wcc, bcc, cat_dst, N_CAT);

        emb_src = emb_dst; pri_src = pri_dst; cat_src = cat_dst;
    }

    // output = concat(item, price) from final layer
    copy_f32<<<4096, 256, 0, stream>>>((float*)d_out, emb_src, NE);
    copy_f32<<<64, 256, 0, stream>>>((float*)d_out + NE, pri_src, (long)N_PRICE * EMB);
}
